// GSModule_17016660427152
// MI455X (gfx1250) — matrix operand order repaired and checked by reading, not compiled, not run on hardware
//
#include <hip/hip_runtime.h>
#include <hip/hip_bf16.h>

typedef __attribute__((ext_vector_type(16))) __bf16 v16bf;
typedef __attribute__((ext_vector_type(8)))  __bf16 v8bf;
typedef __attribute__((ext_vector_type(8)))  float  v8f;

#define B_SZ   1024
#define N_SZ   64
#define D_SZ   512
#define OUT_SZ 2048

#define WPAD   520          // padded W row length in bf16 elems (512 + 8) -> 1040B stride, conflict-free
#define NCOLS  128          // columns per WG = 64 n-branches x 2 output cols
#define LDS_BYTES (NCOLS*WPAD*2 + NCOLS*4)   // W strip (bf16) + c[] (f32) = ~133.6 KB

// ---------- kernel 1: semantic_vec f32 -> bf16 (stays resident in L2, 1MB) ----------
__global__ void cvt_s_bf16(const float* __restrict__ S, __bf16* __restrict__ Sb, int n) {
    int i = blockIdx.x * blockDim.x + threadIdx.x;
    if (i < n) Sb[i] = (__bf16)S[i];
}

// ---------- kernel 2: fused branch-GEMM + bias + relu + sum over branches ----------
__global__ __launch_bounds__(256, 1)
void branch_gemm(const __bf16* __restrict__ Sb,
                 const float*  __restrict__ V,
                 const float*  __restrict__ W,
                 const float*  __restrict__ bias,
                 float*        __restrict__ out)
{
    extern __shared__ char smem[];
    __bf16* Wlds  = (__bf16*)smem;                       // [NCOLS][WPAD] bf16
    float*  c_lds = (float*)(smem + NCOLS * WPAD * 2);   // [NCOLS] f32

    const int tid    = threadIdx.x;
    const int o_base = blockIdx.x * 2;                   // this WG's two output columns

    // ---- Phase 1: stage K=512 x 128-col W strip to LDS as bf16 (W read ONCE from HBM) ----
    // column index j = o_idx*64 + n  => the 64 branch-columns per output col are contiguous
    {
        const int j  = tid >> 1;
        const int kh = (tid & 1) * 256;
        const int n  = j & 63;
        const int o  = o_base + (j >> 6);
        const float* src = W + ((size_t)(n * OUT_SZ + o)) * D_SZ;
        __bf16* dst = Wlds + j * WPAD;
        #pragma unroll 4
        for (int k = kh; k < kh + 256; k += 8) {
            float4 a = *(const float4*)(src + k);
            float4 b = *(const float4*)(src + k + 4);
            v8bf p;
            p[0] = (__bf16)a.x; p[1] = (__bf16)a.y; p[2] = (__bf16)a.z; p[3] = (__bf16)a.w;
            p[4] = (__bf16)b.x; p[5] = (__bf16)b.y; p[6] = (__bf16)b.z; p[7] = (__bf16)b.w;
            *(v8bf*)(dst + k) = p;
        }
    }
    __syncthreads();

    // ---- Phase 2: c[j] = bias[n,o] - v[n]·W[n,o,:]  (batch-independent constant) ----
    if (tid < NCOLS) {
        const int j = tid;
        const int n = j & 63;
        const int o = o_base + (j >> 6);
        const __bf16* wr = Wlds + j * WPAD;
        const float*  vr = V + n * D_SZ;
        float acc = bias[n * OUT_SZ + o];
        #pragma unroll 4
        for (int k = 0; k < D_SZ; k += 4) {
            float4 vv = *(const float4*)(vr + k);
            acc -= (float)wr[k]     * vv.x + (float)wr[k + 1] * vv.y +
                   (float)wr[k + 2] * vv.z + (float)wr[k + 3] * vv.w;
        }
        c_lds[j] = acc;
    }
    __syncthreads();

    // ---- Phase 3: GEMM over all 1024 batch rows, 8 tiles of 128 ----
    const int wave = tid >> 5, lane = tid & 31;
    const int wrow = wave & 3;          // 4 row-waves  -> 32 rows each
    const int wcol = wave >> 2;         // 2 col-waves  -> 64 cols each (= all n for one o)
    const int lm   = lane & 15;         // position within 16-lane half
    const int lh   = lane >> 4;         // which half (selects K sub-range per ISA layout)

    for (int bt = 0; bt < 8; ++bt) {
        const int r0 = bt * 128 + wrow * 32;

        v8f acc[2][4];
        #pragma unroll
        for (int rt = 0; rt < 2; ++rt)
            #pragma unroll
            for (int ct = 0; ct < 4; ++ct)
                acc[rt][ct] = (v8f){};

        #pragma unroll 2
        for (int kk = 0; kk < D_SZ; kk += 32) {
            // A fragments (16x32 bf16): lane = M row; lane-half selects K = {0..7,16..23} vs {8..15,24..31}
            v16bf afrag[2];
            #pragma unroll
            for (int rt = 0; rt < 2; ++rt) {
                const __bf16* ap = Sb + (size_t)(r0 + rt * 16 + lm) * D_SZ + kk + lh * 8;
                v8bf lo = *(const v8bf*)ap;
                v8bf hi = *(const v8bf*)(ap + 16);
                v16bf f;
                #pragma unroll
                for (int i = 0; i < 8; ++i) { f[i] = lo[i]; f[i + 8] = hi[i]; }
                afrag[rt] = f;
            }
            // B fragments (32x16 bf16) from LDS: lane = N col; half selects K = {0..7,16..23} vs {8..15,24..31}
            v16bf bfrag[4];
            #pragma unroll
            for (int ct = 0; ct < 4; ++ct) {
                const __bf16* bp = Wlds + (size_t)(wcol * 64 + ct * 16 + lm) * WPAD + kk + lh * 8;
                v8bf lo = *(const v8bf*)bp;
                v8bf hi = *(const v8bf*)(bp + 16);
                v16bf f;
                #pragma unroll
                for (int i = 0; i < 8; ++i) { f[i] = lo[i]; f[i + 8] = hi[i]; }
                bfrag[ct] = f;
            }
            #pragma unroll
            for (int rt = 0; rt < 2; ++rt)
                #pragma unroll
                for (int ct = 0; ct < 4; ++ct)
                    acc[rt][ct] = __builtin_amdgcn_wmma_f32_16x16x32_bf16(
                        false, afrag[rt], false, bfrag[ct],
                        (short)0, acc[rt][ct], false, false);
        }

        // ---- Epilogue: +c, relu, sum over this wave's 64 columns (= all n), store out[b,o] ----
        // C layout: VGPR r holds row r (lanes 0-15) / row r+8 (lanes 16-31); lane%16 = column.
        #pragma unroll
        for (int rt = 0; rt < 2; ++rt) {
            #pragma unroll
            for (int vr = 0; vr < 8; ++vr) {
                float s = 0.f;
                #pragma unroll
                for (int ct = 0; ct < 4; ++ct) {
                    const int j = wcol * 64 + ct * 16 + lm;
                    float x = acc[rt][ct][vr] + c_lds[j];
                    s += fmaxf(x, 0.f);
                }
                // reduce across the 16 columns held by this lane-half (xor<16 keeps halves separate)
                s += __shfl_xor(s, 1, 32);
                s += __shfl_xor(s, 2, 32);
                s += __shfl_xor(s, 4, 32);
                s += __shfl_xor(s, 8, 32);
                if (lm == 0) {
                    const int row = r0 + rt * 16 + vr + lh * 8;
                    out[(size_t)row * OUT_SZ + o_base + wcol] = s;
                }
            }
        }
    }
}

extern "C" void kernel_launch(void* const* d_in, const int* in_sizes, int n_in,
                              void* d_out, int out_size, void* d_ws, size_t ws_size,
                              hipStream_t stream) {
    const float* S    = (const float*)d_in[0];   // [B, D]
    const float* V    = (const float*)d_in[1];   // [N, D]
    const float* W    = (const float*)d_in[2];   // [N, OUT, D]
    const float* bias = (const float*)d_in[3];   // [N, OUT]
    float* out = (float*)d_out;                  // [B, OUT]
    __bf16* Sb = (__bf16*)d_ws;                  // 1 MB scratch: S in bf16

    (void)in_sizes; (void)n_in; (void)out_size; (void)ws_size;

    hipFuncSetAttribute(reinterpret_cast<const void*>(branch_gemm),
                        hipFuncAttributeMaxDynamicSharedMemorySize, LDS_BYTES);

    const int n = B_SZ * D_SZ;
    cvt_s_bf16<<<(n + 255) / 256, 256, 0, stream>>>(S, Sb, n);
    branch_gemm<<<OUT_SZ / 2, 256, LDS_BYTES, stream>>>(Sb, V, W, bias, out);
}
